// ParallelCausalSelfAttentionLayer_49529562858328
// MI455X (gfx1250) — matrix operand order repaired and checked by reading, not compiled, not run on hardware
//
#include <hip/hip_runtime.h>
#include <hip/hip_bf16.h>

// MI455X / gfx1250, wave32. bf16 WMMA (v_wmma_f32_16x16x32_bf16) with fp32
// accumulate for all GEMMs; fp32 weights converted to bf16 on the fly.
// GEMM: double-buffered LDS tiles, A-tile copied with CDNA5 async
// global->LDS (ASYNCcnt) when the builtin is available.
//
// Workspace layout (bytes, assumes ws_size >= 56 MiB):
//   [ 0MB) xn   : bf16 [2048][4096]         (post-LN activations)
//   [16MB) q    : bf16 [32][2048][128]      (head-major, rope'd in place)
//   [32MB) k    : bf16 [ 8][2048][128]      (head-major, rope'd in place)
//   [36MB) vT   : bf16 [ 8][128][2048]      (d-major => contiguous B-frags in PV)
//   [40MB) attn : bf16 [2048][4096]         (attention out, row-major A for WO gemm)

#include <stdint.h>

typedef __bf16 bf16;
typedef __attribute__((ext_vector_type(8)))  bf16  v8bf;
typedef __attribute__((ext_vector_type(16))) bf16  v16bf;
typedef __attribute__((ext_vector_type(8)))  float v8f;
typedef __attribute__((ext_vector_type(4)))  float v4f;
typedef __attribute__((ext_vector_type(4)))  int   v4i;

#if defined(__HIP_DEVICE_COMPILE__) && \
    __has_builtin(__builtin_amdgcn_global_load_async_to_lds_b128) && \
    __has_builtin(__builtin_amdgcn_s_wait_asynccnt)
#define USE_ASYNC_LDS 1
#else
#define USE_ASYNC_LDS 0
#endif

static __device__ __forceinline__ v16bf bfcat(v8bf lo, v8bf hi) {
  return __builtin_shufflevector(lo, hi, 0,1,2,3,4,5,6,7,8,9,10,11,12,13,14,15);
}
static __device__ __forceinline__ v8f wmma_bf16(v16bf a, v16bf b, v8f c) {
  // (neg_a, A, neg_b, B, c_mod, C, reuse_a, reuse_b)
  return __builtin_amdgcn_wmma_f32_16x16x32_bf16(false, a, false, b, (short)0, c,
                                                 false, false);
}

#if USE_ASYNC_LDS
// 32B global->LDS async copy (2 x b128). Builtin prototype (from clang):
//   (int4 AS1*, int4 AS3*, const int offset, const int cpol)
// Generic->AS casts go through integers; generic LDS addresses carry the
// LDS offset in the low 32 bits.
static __device__ __forceinline__ void async_copy_32B(const bf16* g, bf16* l) {
  typedef __attribute__((address_space(1))) v4i* gp_t;
  typedef __attribute__((address_space(3))) v4i* lp_t;
  gp_t gp = (gp_t)(uintptr_t)g;
  lp_t lp = (lp_t)(unsigned int)(uintptr_t)l;
  __builtin_amdgcn_global_load_async_to_lds_b128(gp,     lp,     0, 0);
  __builtin_amdgcn_global_load_async_to_lds_b128(gp + 1, lp + 1, 0, 0);
}
#endif

// ---------------------------------------------------------------- LayerNorm
__global__ __launch_bounds__(256)
void ln_kernel(const float* __restrict__ x, const float* __restrict__ w,
               const float* __restrict__ b, bf16* __restrict__ xn)
{
  __shared__ float red[256];
  const int row = blockIdx.x, tid = threadIdx.x;
  const float* xr = x + (size_t)row * 4096;
  float vals[16];
  float s = 0.f;
  #pragma unroll
  for (int i = 0; i < 16; ++i) { vals[i] = xr[tid + (i << 8)]; s += vals[i]; }
  red[tid] = s; __syncthreads();
  for (int off = 128; off > 0; off >>= 1) {
    if (tid < off) red[tid] += red[tid + off];
    __syncthreads();
  }
  const float mu = red[0] * (1.f / 4096.f);
  __syncthreads();
  float s2 = 0.f;
  #pragma unroll
  for (int i = 0; i < 16; ++i) { const float d = vals[i] - mu; s2 += d * d; }
  red[tid] = s2; __syncthreads();
  for (int off = 128; off > 0; off >>= 1) {
    if (tid < off) red[tid] += red[tid + off];
    __syncthreads();
  }
  const float rstd = rsqrtf(red[0] * (1.f / 4096.f) + 1e-5f);
  bf16* xo = xn + (size_t)row * 4096;
  #pragma unroll
  for (int i = 0; i < 16; ++i) {
    const int c = tid + (i << 8);
    xo[c] = (bf16)((vals[i] - mu) * rstd * w[c] + b[c]);
  }
}

// --------------------------------------------------------------------- GEMM
// C[M,N] = A[M,K](bf16) @ W[K,N](fp32 -> bf16).
// Block tile 128x128, 8 waves (4x2), wave tile 32x64 = 2x4 WMMA tiles.
// Double-buffered LDS; one barrier per 32-deep K step.
// MODE 0: fp32 row-major out.  MODE 1: bf16 [head][s][128].  MODE 2: bf16 [head][d][s].
template<int MODE>
__global__ __launch_bounds__(256)
void gemm_kernel(const bf16* __restrict__ A, const float* __restrict__ W,
                 void* __restrict__ Cout, int M, int N, int K)
{
  __shared__ __attribute__((aligned(16))) bf16 At[2][128][40]; // [buf][row][k]
  __shared__ __attribute__((aligned(16))) bf16 Bt[2][128][40]; // [buf][col][k]
  const int tid = threadIdx.x;
  const int wid = tid >> 5, lane = tid & 31;
  const int wm = wid >> 1, wn = wid & 1;
  const int h = lane >> 4, ln = lane & 15;
  const int m0 = blockIdx.y * 128, n0 = blockIdx.x * 128;
  const int arow = tid >> 1, acol = (tid & 1) * 16;
  const int bk = tid >> 3,  bn  = (tid & 7) * 16;

  const bf16*  agBase = A + (size_t)(m0 + arow) * K + acol;
  const float* wgBase = W + (size_t)bk * N + (n0 + bn);

  v8f acc[2][4] = {};
#if !USE_ASYNC_LDS
  v8bf aR0, aR1;
#endif
  float bR[16];

  // ---- prologue: stage tile 0 into buffer 0
#if USE_ASYNC_LDS
  async_copy_32B(agBase, &At[0][arow][acol]);
#else
  aR0 = *(const v8bf*)agBase;
  aR1 = *(const v8bf*)(agBase + 8);
  *(v8bf*)&At[0][arow][acol]     = aR0;
  *(v8bf*)&At[0][arow][acol + 8] = aR1;
#endif
  {
    #pragma unroll
    for (int i = 0; i < 4; ++i) {
      v4f w4 = *(const v4f*)(wgBase + i * 4);
      #pragma unroll
      for (int j = 0; j < 4; ++j) bR[i * 4 + j] = w4[j];
    }
    #pragma unroll
    for (int i = 0; i < 16; ++i) Bt[0][bn + i][bk] = (bf16)bR[i];
  }

  const int nsteps = K >> 5;
  int buf = 0;
  for (int s = 0; s < nsteps; ++s, buf ^= 1) {
#if USE_ASYNC_LDS
    __builtin_amdgcn_s_wait_asynccnt(0);   // our async A-tile stores landed
#endif
    __syncthreads();                       // tile[buf] visible to all waves

    const bool more = (s + 1) < nsteps;
    if (more) {
      const int k1 = (s + 1) << 5;
#if USE_ASYNC_LDS
      async_copy_32B(agBase + k1, &At[buf ^ 1][arow][acol]);
#else
      aR0 = *(const v8bf*)(agBase + k1);
      aR1 = *(const v8bf*)(agBase + k1 + 8);
#endif
      const float* wg = wgBase + (size_t)k1 * N;
      if (s + 2 < nsteps)
        __builtin_prefetch((const void*)(wg + (size_t)32 * N), 0, 0);
      #pragma unroll
      for (int i = 0; i < 4; ++i) {
        v4f w4 = *(const v4f*)(wg + i * 4);
        #pragma unroll
        for (int j = 0; j < 4; ++j) bR[i * 4 + j] = w4[j];
      }
    }

    // ---- compute from tile[buf]
    v16bf af[2], bfrag[4];
    #pragma unroll
    for (int ms = 0; ms < 2; ++ms) {
      const bf16* ap = &At[buf][wm * 32 + ms * 16 + ln][0];
      af[ms] = bfcat(*(const v8bf*)(ap + 8 * h), *(const v8bf*)(ap + 16 + 8 * h));
    }
    #pragma unroll
    for (int ns = 0; ns < 4; ++ns) {
      const bf16* bp = &Bt[buf][wn * 64 + ns * 16 + ln][8 * h];
      bfrag[ns] = bfcat(*(const v8bf*)bp, *(const v8bf*)(bp + 16));
    }
    #pragma unroll
    for (int ms = 0; ms < 2; ++ms)
      #pragma unroll
      for (int ns = 0; ns < 4; ++ns)
        acc[ms][ns] = wmma_bf16(af[ms], bfrag[ns], acc[ms][ns]);

    // ---- stage tile[buf^1] (safe: last read before this step's barrier)
    if (more) {
#if !USE_ASYNC_LDS
      *(v8bf*)&At[buf ^ 1][arow][acol]     = aR0;
      *(v8bf*)&At[buf ^ 1][arow][acol + 8] = aR1;
#endif
      #pragma unroll
      for (int i = 0; i < 16; ++i) Bt[buf ^ 1][bn + i][bk] = (bf16)bR[i];
    }
  }

  #pragma unroll
  for (int ms = 0; ms < 2; ++ms)
    #pragma unroll
    for (int ns = 0; ns < 4; ++ns)
      #pragma unroll
      for (int r = 0; r < 8; ++r) {
        const int row = m0 + wm * 32 + ms * 16 + r + 8 * h;
        const int col = n0 + wn * 64 + ns * 16 + ln;
        const float val = acc[ms][ns][r];
        if (MODE == 0) {
          ((float*)Cout)[(size_t)row * N + col] = val;
        } else if (MODE == 1) {
          const int head = col >> 7, d = col & 127;
          ((bf16*)Cout)[((size_t)head * M + row) * 128 + d] = (bf16)val;
        } else {
          const int head = col >> 7, d = col & 127;
          ((bf16*)Cout)[((size_t)head * 128 + d) * M + row] = (bf16)val;
        }
      }
}

// --------------------------------------------------------------------- RoPE
// In-place rotary on [nheads][S][128]; one thread per (s, head, pair d).
__global__ __launch_bounds__(64)
void rope_kernel(bf16* __restrict__ buf, int S)
{
  const int s = blockIdx.x, head = blockIdx.y, d = threadIdx.x; // d in 0..63
  bf16* p = buf + ((size_t)head * S + s) * 128;
  const float inv = __expf(-((float)(2 * d) * (1.f / 128.f)) * __logf(10000.f));
  const float fr = (float)s * inv;
  const float c = __cosf(fr), sn = __sinf(fr);
  const float x1 = (float)p[d], x2 = (float)p[d + 64];
  p[d]      = (bf16)(x1 * c - x2 * sn);
  p[d + 64] = (bf16)(x2 * c + x1 * sn);
}

// ---------------------------------------------------------- Flash attention
// grid = (S/256, 32 heads), 8 waves, one wave per 32 query rows.
__global__ __launch_bounds__(256)
void attn_kernel(const bf16* __restrict__ q, const bf16* __restrict__ k,
                 const bf16* __restrict__ vT, bf16* __restrict__ attn)
{
  const int S = 2048;
  __shared__ __attribute__((aligned(16))) bf16 Plds[8][32][40]; // per wave P tile
  const int head = blockIdx.y;
  const int kvh = head >> 2;                    // GQA: 4 q-heads per kv-head
  const int tid = threadIdx.x, wid = tid >> 5, lane = tid & 31;
  const int h = lane >> 4, ln = lane & 15;
  const int qbase = blockIdx.x * 256 + wid * 32;

  const bf16* qh = q  + (size_t)head * S * 128;
  const bf16* kh = k  + (size_t)kvh  * S * 128;
  const bf16* vh = vT + (size_t)kvh  * 128 * S;

  v8f accO[2][8] = {};
  float mi[2][8], li[2][8];
  #pragma unroll
  for (int a = 0; a < 2; ++a)
    #pragma unroll
    for (int r = 0; r < 8; ++r) { mi[a][r] = -3.0e38f; li[a][r] = 0.f; }

  const float scale = 0.08838834764831845f;     // 1/sqrt(128)
  const int nsteps = (qbase >> 5) + 1;          // causal: keys <= qbase+31

  for (int t = 0; t < nsteps; ++t) {
    const int j0 = t << 5;
    // ---- S = Q @ K^T over D=128 (4 WMMA K-steps of 32)
    v8f accS[2][2] = {};
    #pragma unroll
    for (int kk = 0; kk < 4; ++kk) {
      const int ko = kk * 32;
      v16bf aq[2], bk2[2];
      #pragma unroll
      for (int ms = 0; ms < 2; ++ms) {
        const bf16* qp = qh + (size_t)(qbase + ms * 16 + ln) * 128 + ko;
        aq[ms] = bfcat(*(const v8bf*)(qp + 8 * h), *(const v8bf*)(qp + 16 + 8 * h));
      }
      #pragma unroll
      for (int ns = 0; ns < 2; ++ns) {
        const bf16* kp = kh + (size_t)(j0 + ns * 16 + ln) * 128 + ko + 8 * h;
        bk2[ns] = bfcat(*(const v8bf*)kp, *(const v8bf*)(kp + 16));
      }
      #pragma unroll
      for (int ms = 0; ms < 2; ++ms)
        #pragma unroll
        for (int ns = 0; ns < 2; ++ns)
          accS[ms][ns] = wmma_bf16(aq[ms], bk2[ns], accS[ms][ns]);
    }
    // ---- online softmax (row lives on 16 lanes sharing h; reduce with xor shfl)
    #pragma unroll
    for (int ms = 0; ms < 2; ++ms) {
      #pragma unroll
      for (int r = 0; r < 8; ++r) {
        const int row = qbase + ms * 16 + r + 8 * h;
        float s0 = accS[ms][0][r] * scale;
        float s1 = accS[ms][1][r] * scale;
        if (j0 + ln > row)      s0 = -3.0e38f;
        if (j0 + 16 + ln > row) s1 = -3.0e38f;
        float mx = fmaxf(s0, s1);
        mx = fmaxf(mx, __shfl_xor(mx, 1, 32));
        mx = fmaxf(mx, __shfl_xor(mx, 2, 32));
        mx = fmaxf(mx, __shfl_xor(mx, 4, 32));
        mx = fmaxf(mx, __shfl_xor(mx, 8, 32));
        const float mnew  = fmaxf(mi[ms][r], mx);
        const float alpha = __expf(mi[ms][r] - mnew);
        const float p0 = __expf(s0 - mnew);
        const float p1 = __expf(s1 - mnew);
        float rs = p0 + p1;
        rs += __shfl_xor(rs, 1, 32);
        rs += __shfl_xor(rs, 2, 32);
        rs += __shfl_xor(rs, 4, 32);
        rs += __shfl_xor(rs, 8, 32);
        li[ms][r] = li[ms][r] * alpha + rs;
        mi[ms][r] = mnew;
        #pragma unroll
        for (int d8 = 0; d8 < 8; ++d8) accO[ms][d8][r] *= alpha;
        // C-layout -> LDS so we can reload P in A-fragment layout
        Plds[wid][ms * 16 + r + 8 * h][ln]      = (bf16)p0;
        Plds[wid][ms * 16 + r + 8 * h][16 + ln] = (bf16)p1;
      }
    }
    __builtin_amdgcn_wave_barrier();  // wave-internal LDS RAW; DS is in-order

    // ---- O += P @ V  (V transposed in ws: contiguous along keys)
    v16bf bv[8];
    #pragma unroll
    for (int d8 = 0; d8 < 8; ++d8) {
      const bf16* vp = vh + (size_t)(d8 * 16 + ln) * S + j0 + 8 * h;
      bv[d8] = bfcat(*(const v8bf*)vp, *(const v8bf*)(vp + 16));
    }
    #pragma unroll
    for (int ms = 0; ms < 2; ++ms) {
      const bf16* pp = &Plds[wid][ms * 16 + ln][0];
      const v16bf ap = bfcat(*(const v8bf*)(pp + 8 * h),
                             *(const v8bf*)(pp + 16 + 8 * h));
      #pragma unroll
      for (int d8 = 0; d8 < 8; ++d8)
        accO[ms][d8] = wmma_bf16(ap, bv[d8], accO[ms][d8]);
    }
    __builtin_amdgcn_wave_barrier();
  }

  #pragma unroll
  for (int ms = 0; ms < 2; ++ms)
    #pragma unroll
    for (int d8 = 0; d8 < 8; ++d8)
      #pragma unroll
      for (int r = 0; r < 8; ++r) {
        const int row = qbase + ms * 16 + r + 8 * h;
        const int d = d8 * 16 + ln;
        attn[(size_t)row * 4096 + head * 128 + d] =
            (bf16)(accO[ms][d8][r] / li[ms][r]);
      }
}

// ------------------------------------------------------------------- launch
extern "C" void kernel_launch(void* const* d_in, const int* in_sizes, int n_in,
                              void* d_out, int out_size, void* d_ws, size_t ws_size,
                              hipStream_t stream)
{
  (void)in_sizes; (void)n_in; (void)out_size; (void)ws_size;
  const float* hs  = (const float*)d_in[0];
  const float* lnw = (const float*)d_in[1];
  const float* lnb = (const float*)d_in[2];
  const float* wq  = (const float*)d_in[3];
  const float* wk  = (const float*)d_in[4];
  const float* wv  = (const float*)d_in[5];
  const float* wo  = (const float*)d_in[6];
  float* out = (float*)d_out;

  const int S = 2048, H = 4096;
  char* ws = (char*)d_ws;
  bf16* xn   = (bf16*)(ws);
  bf16* qb   = (bf16*)(ws + (size_t)16 * 1024 * 1024);
  bf16* kb   = (bf16*)(ws + (size_t)32 * 1024 * 1024);
  bf16* vT   = (bf16*)(ws + (size_t)36 * 1024 * 1024);
  bf16* attn = (bf16*)(ws + (size_t)40 * 1024 * 1024);

  ln_kernel<<<S, 256, 0, stream>>>(hs, lnw, lnb, xn);
  gemm_kernel<1><<<dim3(32, 16), 256, 0, stream>>>(xn, wq, (void*)qb, S, 4096, H);
  gemm_kernel<1><<<dim3( 8, 16), 256, 0, stream>>>(xn, wk, (void*)kb, S, 1024, H);
  gemm_kernel<2><<<dim3( 8, 16), 256, 0, stream>>>(xn, wv, (void*)vT, S, 1024, H);
  rope_kernel<<<dim3(S, 32), 64, 0, stream>>>(qb, S);
  rope_kernel<<<dim3(S,  8), 64, 0, stream>>>(kb, S);
  attn_kernel<<<dim3(8, 32), 256, 0, stream>>>(qb, kb, vT, attn);
  gemm_kernel<0><<<dim3(32, 16), 256, 0, stream>>>(attn, wo, (void*)out, S, 4096, 4096);
}
